// GAT_43568148251054
// MI455X (gfx1250) — matrix operand order repaired and checked by reading, not compiled, not run on hardware
//
#include <hip/hip_runtime.h>
#include <hip/hip_bf16.h>
#include <math.h>

// ---------------- types for WMMA ----------------
typedef _Float16 v16h __attribute__((ext_vector_type(16)));
typedef _Float16 v8h  __attribute__((ext_vector_type(8)));
typedef float    v8f  __attribute__((ext_vector_type(8)));

// ---------------- small helpers ----------------
__device__ __forceinline__ void atomicMaxF(float* addr, float val) {
    // IEEE trick: positive floats ordered as signed ints, negative as reversed unsigned
    if (val >= 0.0f) atomicMax((int*)addr, __float_as_int(val));
    else             atomicMin((unsigned int*)addr, __float_as_uint(val));
}

__device__ __forceinline__ void edge_sd(const int* __restrict__ ei, int E, int e,
                                        int& s, int& d) {
    if (e < E) { s = ei[e]; d = ei[E + e]; }
    else       { s = e - E; d = e - E; }          // appended self loops
}

// ---------------- utility kernels ----------------
__global__ void gat_fill_f32(float* __restrict__ p, float v, int n) {
    int t = blockIdx.x * blockDim.x + threadIdx.x;
    if (t < n) p[t] = v;
}

__global__ void gat_f32_to_f16(const float* __restrict__ in, _Float16* __restrict__ out, int n) {
    int t = blockIdx.x * blockDim.x + threadIdx.x;
    if (t < n) out[t] = (_Float16)in[t];
}

// Pack W [K, Ncols] (f32 row-major) into WMMA B-fragment order:
// linear t = ((nt*KC + kc)*32 + lane)*16 + j  ->  element (k = kc*32 + (j&7) + 8*laneHi + 16*(j>>3), n = nt*16 + lane&15)
__global__ void gat_pack_b(const float* __restrict__ W, _Float16* __restrict__ Bp,
                           int Ncols, int KC, int total) {
    int t = blockIdx.x * blockDim.x + threadIdx.x;
    if (t >= total) return;
    int j    = t & 15;
    int lane = (t >> 4) & 31;
    int rest = t >> 9;
    int kc   = rest % KC;
    int nt   = rest / KC;
    int k = kc * 32 + (j & 7) + ((lane >> 4) << 3) + ((j >> 3) << 4);
    int n = nt * 16 + (lane & 15);
    Bp[t] = (_Float16)W[(size_t)k * Ncols + n];
}

// ---------------- WMMA GEMM: C[M,N] = A[M,K] * B ----------------
// A: f16 row-major. Bp: pre-packed fragments. One wave per 16x16 C tile.
__global__ __launch_bounds__(256)
void gat_gemm_wmma(const _Float16* __restrict__ A, const _Float16* __restrict__ Bp,
                   float* __restrict__ C, int M, int K, int Ntiles, int tilesTotal) {
    int wave = blockIdx.x * (blockDim.x >> 5) + (threadIdx.x >> 5);
    if (wave >= tilesTotal) return;                 // uniform per wave: EXEC stays all-1s
    int lane   = threadIdx.x & 31;
    int laneHi = lane >> 4;                         // 0 or 1
    int mt = wave / Ntiles;
    int nt = wave % Ntiles;
    int KC = K >> 5;
    int Ncols = Ntiles * 16;

    const _Float16* arow  = A  + (size_t)(mt * 16 + (lane & 15)) * K;
    const _Float16* bbase = Bp + (size_t)nt * KC * (32 * 16);

    v8f acc = {};
    for (int kc = 0; kc < KC; ++kc) {
        // A 16x32 f16 fragment: lanes 0-15 hold K 0..7 & 16..23, lanes 16-31 hold K 8..15 & 24..31
        const _Float16* ap = arow + kc * 32 + laneHi * 8;
        v8h alo = *(const v8h*)(ap);
        v8h ahi = *(const v8h*)(ap + 16);
        v16h a = __builtin_shufflevector(alo, ahi, 0,1,2,3,4,5,6,7,8,9,10,11,12,13,14,15);
        // B fragment: 16 contiguous packed halves per lane
        v16h b = *(const v16h*)(bbase + (size_t)(kc * 32 + lane) * 16);
        acc = __builtin_amdgcn_wmma_f32_16x16x32_f16(false, a, false, b,
                                                     (short)0, acc, false, false);
    }
    // C/D layout: VGPR r -> row r + 8*laneHi, col = lane&15
    float* crow = C + (size_t)(mt * 16 + laneHi * 8) * Ncols + nt * 16 + (lane & 15);
    #pragma unroll
    for (int r = 0; r < 8; ++r)
        crow[(size_t)r * Ncols] = acc[r];
}

// ---------------- attention scores: a_src/a_dst [N,H] ----------------
__global__ void gat_att_scores(const float* __restrict__ xl,
                               const float* __restrict__ att_s,
                               const float* __restrict__ att_d,
                               float* __restrict__ asrc, float* __restrict__ adst,
                               int N, int H, int Cc) {
    int t = blockIdx.x * blockDim.x + threadIdx.x;
    if (t >= N * H) return;
    int n = t / H, h = t % H;
    const float* xp = xl + ((size_t)n * H + h) * Cc;
    const float* as = att_s + h * Cc;
    const float* ad = att_d + h * Cc;
    float ss = 0.f, sd = 0.f;
    for (int c = 0; c < Cc; ++c) { float v = xp[c]; ss += v * as[c]; sd += v * ad[c]; }
    asrc[t] = ss; adst[t] = sd;
}

// ---------------- edge pass 1: segment max of leaky-relu scores ----------------
__global__ void gat_edge_max(const int* __restrict__ ei, int E, int ET,
                             const float* __restrict__ asrc, const float* __restrict__ adst,
                             float* __restrict__ segmax, int H) {
    int t = blockIdx.x * blockDim.x + threadIdx.x;
    if (t >= ET * H) return;
    int e = t / H, h = t % H;
    int s, d; edge_sd(ei, E, e, s, d);
    float v = asrc[s * H + h] + adst[d * H + h];
    v = v >= 0.f ? v : 0.2f * v;
    atomicMaxF(&segmax[d * H + h], v);
}

// ---------------- edge pass 2: exp + segment sum ----------------
__global__ void gat_edge_expsum(const int* __restrict__ ei, int E, int ET,
                                const float* __restrict__ asrc, const float* __restrict__ adst,
                                const float* __restrict__ segmax,
                                float* __restrict__ exbuf, float* __restrict__ segsum, int H) {
    int t = blockIdx.x * blockDim.x + threadIdx.x;
    if (t >= ET * H) return;
    int e = t / H, h = t % H;
    int s, d; edge_sd(ei, E, e, s, d);
    float v = asrc[s * H + h] + adst[d * H + h];
    v = v >= 0.f ? v : 0.2f * v;
    float ex = __expf(v - segmax[d * H + h]);
    exbuf[t] = ex;
    atomicAdd(&segsum[d * H + h], ex);
}

// ---------------- edge pass 3: alpha-weighted scatter ----------------
// thread handles 4 channels of one (edge, head)
__global__ void gat_edge_scatter(const int* __restrict__ ei, int E, int ET,
                                 const float* __restrict__ exbuf,
                                 const float* __restrict__ segsum,
                                 const float* __restrict__ xl,
                                 float* __restrict__ acc, int H, int Cc) {
    int groups = Cc >> 2;
    int total  = ET * H * groups;
    int t = blockIdx.x * blockDim.x + threadIdx.x;
    if (t >= total) return;
    int g   = t % groups;
    int rem = t / groups;
    int h   = rem % H;
    int e   = rem / H;
    int s, d; edge_sd(ei, E, e, s, d);
    float alpha = exbuf[(size_t)e * H + h] / (segsum[d * H + h] + 1e-16f);
    const float4 xv = *(const float4*)(xl + ((size_t)s * H + h) * Cc + g * 4);
    float* op = acc + ((size_t)d * H + h) * Cc + g * 4;
    atomicAdd(op + 0, xv.x * alpha);
    atomicAdd(op + 1, xv.y * alpha);
    atomicAdd(op + 2, xv.z * alpha);
    atomicAdd(op + 3, xv.w * alpha);
}

// ---------------- bias + ELU -> f16 (feeds GEMM2) ----------------
__global__ void gat_bias_elu_f16(const float* __restrict__ acc, const float* __restrict__ b,
                                 _Float16* __restrict__ out, int total, int F) {
    int t = blockIdx.x * blockDim.x + threadIdx.x;
    if (t >= total) return;
    float v = acc[t] + b[t % F];
    v = v > 0.f ? v : (__expf(v) - 1.f);
    out[t] = (_Float16)v;
}

// ---------------- final bias add (in place on d_out) ----------------
__global__ void gat_add_bias(float* __restrict__ out, const float* __restrict__ b,
                             int total, int F) {
    int t = blockIdx.x * blockDim.x + threadIdx.x;
    if (t >= total) return;
    out[t] += b[t % F];
}

// =====================================================================
extern "C" void kernel_launch(void* const* d_in, const int* in_sizes, int n_in,
                              void* d_out, int out_size, void* d_ws, size_t ws_size,
                              hipStream_t stream) {
    const float* x        = (const float*)d_in[0];
    const int*   ei       = (const int*)  d_in[1];
    const float* W1       = (const float*)d_in[2];
    const float* att_src1 = (const float*)d_in[3];
    const float* att_dst1 = (const float*)d_in[4];
    const float* b1       = (const float*)d_in[5];
    const float* W2       = (const float*)d_in[6];
    const float* att_src2 = (const float*)d_in[7];
    const float* att_dst2 = (const float*)d_in[8];
    const float* b2       = (const float*)d_in[9];
    float* out = (float*)d_out;

    const int F_in = 128, H1 = 8, C1 = 16, F_mid = 128, H2 = 1, C2 = 64;
    const int N  = in_sizes[0] / F_in;      // 50000 (multiple of 16)
    const int E  = in_sizes[1] / 2;         // 800000
    const int ET = E + N;                   // + self loops

    // ---- carve workspace ----
    char* w = (char*)d_ws;
    auto carve = [&](size_t bytes) -> char* {
        char* p = w; w += (bytes + 255) & ~(size_t)255; return p;
    };
    _Float16* xh      = (_Float16*)carve((size_t)N * F_in * 2);
    _Float16* w1p     = (_Float16*)carve((size_t)(F_mid/16) * (F_in/32) * 32 * 16 * 2);
    _Float16* w2p     = (_Float16*)carve((size_t)(C2/16)    * (F_mid/32) * 32 * 16 * 2);
    float*    xl1     = (float*)   carve((size_t)N * F_mid * 4);
    float*    asrc1   = (float*)   carve((size_t)N * H1 * 4);
    float*    adst1   = (float*)   carve((size_t)N * H1 * 4);
    float*    segmax1 = (float*)   carve((size_t)N * H1 * 4);
    float*    segsum1 = (float*)   carve((size_t)N * H1 * 4);
    float*    ex1     = (float*)   carve((size_t)ET * H1 * 4);
    float*    acc1    = (float*)   carve((size_t)N * F_mid * 4);
    _Float16* hh      = (_Float16*)carve((size_t)N * F_mid * 2);
    float*    xl2     = (float*)   carve((size_t)N * C2 * 4);
    float*    asrc2   = (float*)   carve((size_t)N * 4);
    float*    adst2   = (float*)   carve((size_t)N * 4);
    float*    segmax2 = (float*)   carve((size_t)N * 4);
    float*    segsum2 = (float*)   carve((size_t)N * 4);
    float*    ex2     = (float*)   carve((size_t)ET * 4);

    const int B = 256;
    auto g = [](int n, int b) { return (n + b - 1) / b; };

    // ---- init ----
    gat_fill_f32<<<g(N*H1, B), B, 0, stream>>>(segmax1, -INFINITY, N*H1);
    gat_fill_f32<<<g(N*H1, B), B, 0, stream>>>(segsum1, 0.f, N*H1);
    gat_fill_f32<<<g(N*F_mid, B), B, 0, stream>>>(acc1, 0.f, N*F_mid);
    gat_fill_f32<<<g(N, B), B, 0, stream>>>(segmax2, -INFINITY, N);
    gat_fill_f32<<<g(N, B), B, 0, stream>>>(segsum2, 0.f, N);
    gat_fill_f32<<<g(N*C2, B), B, 0, stream>>>(out, 0.f, N*C2);

    // ---- layer 1 GEMM: xl1 = x @ W1 (f16 WMMA, f32 accum) ----
    gat_f32_to_f16<<<g(N*F_in, B), B, 0, stream>>>(x, xh, N*F_in);
    {
        int KC = F_in / 32, Ntiles = F_mid / 16;
        int packTotal = Ntiles * KC * 32 * 16;
        gat_pack_b<<<g(packTotal, B), B, 0, stream>>>(W1, w1p, F_mid, KC, packTotal);
        int tiles = (N / 16) * Ntiles;
        gat_gemm_wmma<<<g(tiles, 8), 256, 0, stream>>>(xh, w1p, xl1, N, F_in, Ntiles, tiles);
    }

    // ---- layer 1 attention ----
    gat_att_scores<<<g(N*H1, B), B, 0, stream>>>(xl1, att_src1, att_dst1, asrc1, adst1, N, H1, C1);
    gat_edge_max   <<<g(ET*H1, B), B, 0, stream>>>(ei, E, ET, asrc1, adst1, segmax1, H1);
    gat_edge_expsum<<<g(ET*H1, B), B, 0, stream>>>(ei, E, ET, asrc1, adst1, segmax1, ex1, segsum1, H1);
    gat_edge_scatter<<<g(ET*H1*(C1/4), B), B, 0, stream>>>(ei, E, ET, ex1, segsum1, xl1, acc1, H1, C1);

    // ---- bias + ELU -> f16 ----
    gat_bias_elu_f16<<<g(N*F_mid, B), B, 0, stream>>>(acc1, b1, hh, N*F_mid, F_mid);

    // ---- layer 2 GEMM: xl2 = h @ W2 ----
    {
        int KC = F_mid / 32, Ntiles = C2 / 16;
        int packTotal = Ntiles * KC * 32 * 16;
        gat_pack_b<<<g(packTotal, B), B, 0, stream>>>(W2, w2p, C2, KC, packTotal);
        int tiles = (N / 16) * Ntiles;
        gat_gemm_wmma<<<g(tiles, 8), 256, 0, stream>>>(hh, w2p, xl2, N, F_mid, Ntiles, tiles);
    }

    // ---- layer 2 attention (H=1, C=64), scatter straight into d_out ----
    gat_att_scores<<<g(N*H2, B), B, 0, stream>>>(xl2, att_src2, att_dst2, asrc2, adst2, N, H2, C2);
    gat_edge_max   <<<g(ET*H2, B), B, 0, stream>>>(ei, E, ET, asrc2, adst2, segmax2, H2);
    gat_edge_expsum<<<g(ET*H2, B), B, 0, stream>>>(ei, E, ET, asrc2, adst2, segmax2, ex2, segsum2, H2);
    gat_edge_scatter<<<g(ET*H2*(C2/4), B), B, 0, stream>>>(ei, E, ET, ex2, segsum2, xl2, out, H2, C2);

    // ---- final bias ----
    gat_add_bias<<<g(N*C2, B), B, 0, stream>>>(out, b2, N*C2, C2);
}
